// TableDetect_50294067036755
// MI455X (gfx1250) — matrix operand order repaired and checked by reading, not compiled, not run on hardware
//
#include <hip/hip_runtime.h>

// CDNA5 / gfx1250 implementation: all GEMMs via v_wmma_f32_16x16x32_f16
// (f16 inputs, f32 accumulate). Weights converted once per launch into
// zero-padded f16 layouts in workspace; activations stored as padded f16.

typedef _Float16 v16h __attribute__((ext_vector_type(16)));
typedef float    v8f  __attribute__((ext_vector_type(8)));

union HVec  { uint4 u[2]; v16h h; };
union H4Pk  { _Float16 h[4]; uint2 u; };

#define ACT_NONE 0
#define ACT_RELU 1
#define ACT_SIG  2

#define LPITCH 72   // 64-half chunk row + 8 pad -> conflict-free b128 frag reads

__device__ __forceinline__ float sigf(float x) { return 1.0f / (1.0f + __expf(-x)); }

// Y[M, NT*16] = act( A[M,KP] @ W^T + bias ), W is [NT*16][KP] f16 row-major.
// SRCM: 0 = padded-f16 source (row stride lda), 1 = f32 source (row stride
// lda, real width k_real, zero-fill beyond), 2 = gather mode (5 segments of
// 128 cols from gsrc[idx[m][seg]], zeroed when nf[m]).
// OUTF32: out is float, row stride out_stride, only cols < out_col_off kept.
template<int KP, int NT, int SRCM, bool OUTF32>
__global__ void __launch_bounds__(256)
gemm_wmma(const void* __restrict__ Asrc, int lda, int k_real,
          const _Float16* __restrict__ W, const float* __restrict__ bias,
          void* __restrict__ Out, int out_stride, int out_col_off,
          int act, int M,
          const int* __restrict__ idx, const unsigned char* __restrict__ nf,
          const _Float16* __restrict__ gsrc)
{
  constexpr int CH  = (KP >= 64) ? 64 : KP;   // K-chunk staged per iteration
  constexpr int SUB = CH / 32;                // wmma K-steps per chunk

  __shared__ _Float16 lds[8][16 * LPITCH];    // one 16xCH panel per wave

  const int  tid    = threadIdx.x;
  const int  wave   = tid >> 5;
  const int  lane   = tid & 31;
  const long strip  = (long)blockIdx.x * 8 + wave;
  const bool active = (strip + 1) * 16 <= (long)M;
  const long m0     = active ? strip * 16 : 0;   // clamp: uniform loop counts

  _Float16* slds = lds[wave];

  v8f acc[NT];
  const v8f zro = {0.f,0.f,0.f,0.f,0.f,0.f,0.f,0.f};
  #pragma unroll
  for (int i = 0; i < NT; ++i) acc[i] = zro;

  const int arow = lane & 15;                 // A frag: row within strip
  const int aoff = (lane < 16) ? 0 : 16;      // A frag: K sub-offset
  const int boff = (lane < 16) ? 0 : 16;      // B frag: K sub-offset
  const int bn   = lane & 15;                 // B frag: column within tile

  #pragma unroll 1
  for (int kc = 0; kc < KP; kc += CH) {
    // ---- stage 16 x CH A-chunk into this wave's LDS slice ----
    if (SRCM == 0) {
      constexpr int CPR = CH / 8;             // uint4 chunks per row
      #pragma unroll
      for (int i = 0; i < CH / 16; ++i) {
        int c = lane + 32 * i;
        int row = c / CPR, col = (c % CPR) * 8;
        uint4 v = *(const uint4*)((const _Float16*)Asrc +
                                  (m0 + row) * (long)lda + kc + col);
        *(uint4*)(slds + row * LPITCH + col) = v;
      }
    } else if (SRCM == 1) {
      constexpr int CPR = CH / 4;             // float4 chunks per row
      #pragma unroll
      for (int i = 0; i < CH / 8; ++i) {
        int c = lane + 32 * i;
        int row = c / CPR, col = (c % CPR) * 4;
        int gk = kc + col;
        float4 v = make_float4(0.f, 0.f, 0.f, 0.f);
        if (gk < k_real)
          v = *(const float4*)((const float*)Asrc + (m0 + row) * (long)lda + gk);
        H4Pk p;
        p.h[0] = (_Float16)v.x; p.h[1] = (_Float16)v.y;
        p.h[2] = (_Float16)v.z; p.h[3] = (_Float16)v.w;
        *(uint2*)(slds + row * LPITCH + col) = p.u;
      }
    } else {  // gather: ww[m][k] = nf[m] ? 0 : uu[idx[m][k/128]][k%128]
      constexpr int CPR = CH / 8;
      #pragma unroll
      for (int i = 0; i < CH / 16; ++i) {
        int c = lane + 32 * i;
        int row = c / CPR, col = (c % CPR) * 8;
        int gk  = kc + col;
        int seg = gk >> 7, j = gk & 127;
        long gm = m0 + row;
        uint4 v = make_uint4(0u, 0u, 0u, 0u);
        if (!nf[gm]) {
          long srow = idx[gm * 5 + seg];
          v = *(const uint4*)(gsrc + srow * 128 + j);
        }
        *(uint4*)(slds + row * LPITCH + col) = v;
      }
    }
    __syncthreads();

    // ---- compute: SUB wmma K-steps over all NT column tiles ----
    #pragma unroll
    for (int s = 0; s < SUB; ++s) {
      const int kb = s * 32;
      HVec af;   // A 16x32 f16 fragment (contiguous k = 16h + s) from LDS
      af.u[0] = *(const uint4*)(slds + arow * LPITCH + kb + aoff);
      af.u[1] = *(const uint4*)(slds + arow * LPITCH + kb + aoff + 8);
      #pragma unroll
      for (int nt = 0; nt < NT; ++nt) {
        const _Float16* wp = W + (long)(nt * 16 + bn) * KP + (kc + kb + boff);
        HVec bf;  // B 32x16 fragment: lane holds W[n][k..k+15]
        bf.u[0] = ((const uint4*)wp)[0];
        bf.u[1] = ((const uint4*)wp)[1];
        acc[nt] = __builtin_amdgcn_wmma_f32_16x16x32_f16(
            false, af.h, false, bf.h, (short)0, acc[nt], false, false);
      }
    }
    __syncthreads();
  }

  // ---- epilogue: bias + activation + store ----
  if (active) {
    const int col   = lane & 15;
    const int rbase = (lane < 16) ? 0 : 8;
    #pragma unroll
    for (int nt = 0; nt < NT; ++nt) {
      const int   n  = nt * 16 + col;
      const float bv = bias[n];
      #pragma unroll
      for (int v = 0; v < 8; ++v) {
        float x = acc[nt][v] + bv;
        if (act == ACT_RELU)      x = fmaxf(x, 0.0f);
        else if (act == ACT_SIG)  x = sigf(x);
        const long m = m0 + rbase + v;
        if (OUTF32) {
          if (n < out_col_off)
            ((float*)Out)[m * (long)out_stride + n] = x;
        } else {
          ((_Float16*)Out)[m * (long)out_stride + out_col_off + n] = (_Float16)x;
        }
      }
    }
  }
}

// GRU cell with hh==0 (ITERS==1): gi already contains ww@W_ih^T + b_ih.
// r = sig(gi_r + b_hh_r); z = sig(gi_z + b_hh_z); n = tanh(gi_n + r*b_hh_n);
// bb = (1-z)*n.  gi padded [M,320] (real cols 0..299), bb padded [M,128].
__global__ void gru_pointwise(const _Float16* __restrict__ gi,
                              const float* __restrict__ b_hh,
                              _Float16* __restrict__ bb, int M)
{
  long t = (long)blockIdx.x * blockDim.x + threadIdx.x;
  if (t >= (long)M * 128) return;
  int  j = (int)(t & 127);
  long m = t >> 7;
  float o = 0.0f;
  if (j < 100) {
    float ir = (float)gi[m * 320 + j];
    float iz = (float)gi[m * 320 + 100 + j];
    float in = (float)gi[m * 320 + 200 + j];
    float r  = sigf(ir + b_hh[j]);
    float z  = sigf(iz + b_hh[100 + j]);
    float nn = tanhf(in + r * b_hh[200 + j]);
    o = (1.0f - z) * nn;
  }
  bb[m * 128 + j] = (_Float16)o;
}

// f32 [Nr,Kr] -> zero-padded f16 [Np,Kp]; optional segment remap
// (segIn>0): col k -> seg=k/segOut, j=k%segOut, src col = seg*segIn+j.
__global__ void prep_w(const float* __restrict__ src, _Float16* __restrict__ dst,
                       int Nr, int Np, int Kr, int Kp, int segIn, int segOut)
{
  long t = (long)blockIdx.x * blockDim.x + threadIdx.x;
  if (t >= (long)Np * Kp) return;
  int k = (int)(t % Kp), n = (int)(t / Kp);
  float v = 0.0f;
  if (n < Nr) {
    int sk; bool ok;
    if (segIn > 0) {
      int s = k / segOut, j = k % segOut;
      sk = s * segIn + j;
      ok = (j < segIn) && (sk < Kr);
    } else { sk = k; ok = (k < Kr); }
    if (ok) v = src[(long)n * Kr + sk];
  }
  dst[t] = (_Float16)v;
}

__global__ void prep_b(const float* __restrict__ src, float* __restrict__ dst,
                       int Nr, int Np)
{
  int n = blockIdx.x * blockDim.x + threadIdx.x;
  if (n < Np) dst[n] = (n < Nr) ? src[n] : 0.0f;
}

extern "C" void kernel_launch(void* const* d_in, const int* in_sizes, int n_in,
                              void* d_out, int out_size, void* d_ws, size_t ws_size,
                              hipStream_t stream)
{
  (void)n_in; (void)out_size; (void)ws_size;
  const int M      = in_sizes[1];       // N (indices_not_found length)
  const int strips = M / 16;
  const dim3 GB((strips + 7) / 8), TB(256);

  const int*           idx    = (const int*)d_in[0];
  const unsigned char* nf     = (const unsigned char*)d_in[1];
  const void*          x_emb  = d_in[2];   // [N,300] f32
  const void*          x_pos  = d_in[3];   // [N,8]   f32
  const void*          x_conv = d_in[4];   // [N,256] f32
  // params flat (insertion order): 6..11 A_emb, 12..17 A_pos, 18..23 A_conv,
  // 24,25 A_proj, 26 W_ih 27 W_hh 28 b_ih 29 b_hh, 30..35 BO1, 36..41 BO2,
  // 42,43 C, 44,45 D  (BO2/D/W_hh dead with ITERS==1)

  char* ws = (char*)d_ws;
  size_t off = 0;
  auto carve = [&](size_t bytes) -> void* {
    off = (off + 255) & ~(size_t)255;
    void* p = ws + off; off += bytes; return p;
  };
  _Float16* t0  = (_Float16*)carve((size_t)M * 320 * 2);
  _Float16* t1  = (_Float16*)carve((size_t)M * 320 * 2);
  _Float16* cat = (_Float16*)carve((size_t)M * 384 * 2);  // uu_e|uu_p|uu_c
  _Float16* uu  = (_Float16*)carve((size_t)M * 128 * 2);
  _Float16* gi  = cat;                                    // reuse (320<=384)

  auto WB = [&](int Np, int Kp) { return (_Float16*)carve((size_t)Np * Kp * 2); };
  auto BB = [&](int Np)         { return (float*)carve((size_t)Np * 4); };
  _Float16 *w_e1=WB(320,320), *w_e2=WB(320,320), *w_e3=WB(128,320);
  _Float16 *w_p1=WB(32,32),   *w_p2=WB(32,32),   *w_p3=WB(128,32);
  _Float16 *w_c1=WB(128,256), *w_c2=WB(128,128), *w_c3=WB(128,128);
  _Float16 *w_pr=WB(128,384), *w_ih=WB(320,640);
  _Float16 *w_o1=WB(128,128), *w_o2=WB(128,128), *w_o3=WB(128,128);
  _Float16 *w_cc=WB(128,128);
  float *b_e1=BB(320), *b_e2=BB(320), *b_e3=BB(128);
  float *b_p1=BB(32),  *b_p2=BB(32),  *b_p3=BB(128);
  float *b_c1=BB(128), *b_c2=BB(128), *b_c3=BB(128);
  float *b_pr=BB(128), *b_gi=BB(320);
  float *b_o1=BB(128), *b_o2=BB(128), *b_o3=BB(128), *b_cc=BB(128);

  auto PW = [&](const void* s, _Float16* d, int Nr, int Np, int Kr, int Kp,
                int si, int so) {
    long tot = (long)Np * Kp;
    prep_w<<<dim3((unsigned)((tot + 255) / 256)), TB, 0, stream>>>(
        (const float*)s, d, Nr, Np, Kr, Kp, si, so);
  };
  auto PB = [&](const void* s, float* d, int Nr, int Np) {
    prep_b<<<dim3((unsigned)((Np + 255) / 256)), TB, 0, stream>>>(
        (const float*)s, d, Nr, Np);
  };

  PW(d_in[6],  w_e1, 300,320,300,320, 0,0);   PB(d_in[7],  b_e1, 300,320);
  PW(d_in[8],  w_e2, 300,320,300,320, 0,0);   PB(d_in[9],  b_e2, 300,320);
  PW(d_in[10], w_e3, 100,128,300,320, 0,0);   PB(d_in[11], b_e3, 100,128);
  PW(d_in[12], w_p1,  20, 32,  8, 32, 0,0);   PB(d_in[13], b_p1,  20, 32);
  PW(d_in[14], w_p2,  20, 32, 20, 32, 0,0);   PB(d_in[15], b_p2,  20, 32);
  PW(d_in[16], w_p3, 100,128, 20, 32, 0,0);   PB(d_in[17], b_p3, 100,128);
  PW(d_in[18], w_c1, 100,128,256,256, 0,0);   PB(d_in[19], b_c1, 100,128);
  PW(d_in[20], w_c2, 100,128,100,128, 0,0);   PB(d_in[21], b_c2, 100,128);
  PW(d_in[22], w_c3, 100,128,100,128, 0,0);   PB(d_in[23], b_c3, 100,128);
  PW(d_in[24], w_pr, 100,128,300,384, 100,128); PB(d_in[25], b_pr, 100,128);
  PW(d_in[26], w_ih, 300,320,500,640, 100,128); PB(d_in[28], b_gi, 300,320);
  PW(d_in[30], w_o1, 100,128,100,128, 0,0);   PB(d_in[31], b_o1, 100,128);
  PW(d_in[32], w_o2, 100,128,100,128, 0,0);   PB(d_in[33], b_o2, 100,128);
  PW(d_in[34], w_o3, 100,128,100,128, 0,0);   PB(d_in[35], b_o3, 100,128);
  PW(d_in[42], w_cc, 100,128,100,128, 0,0);   PB(d_in[43], b_cc, 100,128);

  // A_emb chain -> cat[:,0:128]
  gemm_wmma<320,20,1,false><<<GB,TB,0,stream>>>(x_emb,300,300, w_e1,b_e1, t0,320,0,  ACT_RELU,M, nullptr,nullptr,nullptr);
  gemm_wmma<320,20,0,false><<<GB,TB,0,stream>>>(t0,   320,0,   w_e2,b_e2, t1,320,0,  ACT_RELU,M, nullptr,nullptr,nullptr);
  gemm_wmma<320, 8,0,false><<<GB,TB,0,stream>>>(t1,   320,0,   w_e3,b_e3, cat,384,0, ACT_NONE,M, nullptr,nullptr,nullptr);
  // A_pos chain -> cat[:,128:256]
  gemm_wmma< 32, 2,1,false><<<GB,TB,0,stream>>>(x_pos,  8,  8, w_p1,b_p1, t0, 32,0,  ACT_SIG, M, nullptr,nullptr,nullptr);
  gemm_wmma< 32, 2,0,false><<<GB,TB,0,stream>>>(t0,    32,0,   w_p2,b_p2, t1, 32,0,  ACT_RELU,M, nullptr,nullptr,nullptr);
  gemm_wmma< 32, 8,0,false><<<GB,TB,0,stream>>>(t1,    32,0,   w_p3,b_p3, cat,384,128,ACT_NONE,M, nullptr,nullptr,nullptr);
  // A_conv chain -> cat[:,256:384]
  gemm_wmma<256, 8,1,false><<<GB,TB,0,stream>>>(x_conv,256,256,w_c1,b_c1, t0,128,0,  ACT_RELU,M, nullptr,nullptr,nullptr);
  gemm_wmma<128, 8,0,false><<<GB,TB,0,stream>>>(t0,   128,0,   w_c2,b_c2, t1,128,0,  ACT_RELU,M, nullptr,nullptr,nullptr);
  gemm_wmma<128, 8,0,false><<<GB,TB,0,stream>>>(t1,   128,0,   w_c3,b_c3, cat,384,256,ACT_NONE,M, nullptr,nullptr,nullptr);
  // uu = relu(cat @ Wproj^T + b)
  gemm_wmma<384, 8,0,false><<<GB,TB,0,stream>>>(cat,  384,0,   w_pr,b_pr, uu,128,0,  ACT_RELU,M, nullptr,nullptr,nullptr);
  // gi = gather(uu) @ W_ih^T + b_ih   (fused neighbor gather)
  gemm_wmma<640,20,2,false><<<GB,TB,0,stream>>>(nullptr,0,0,   w_ih,b_gi, gi,320,0,  ACT_NONE,M, idx, nf, uu);
  // bb = GRU(gi, hh=0) -> t0
  {
    long tot = (long)M * 128;
    gru_pointwise<<<dim3((unsigned)((tot + 255) / 256)), TB, 0, stream>>>(
        gi, (const float*)d_in[29], t0, M);
  }
  // BO1 sigmoid chain, then ll = oo @ C^T + b  -> d_out [N,100] f32
  gemm_wmma<128, 8,0,false><<<GB,TB,0,stream>>>(t0,128,0, w_o1,b_o1, t1,128,0, ACT_SIG, M, nullptr,nullptr,nullptr);
  gemm_wmma<128, 8,0,false><<<GB,TB,0,stream>>>(t1,128,0, w_o2,b_o2, t0,128,0, ACT_SIG, M, nullptr,nullptr,nullptr);
  gemm_wmma<128, 8,0,false><<<GB,TB,0,stream>>>(t0,128,0, w_o3,b_o3, t1,128,0, ACT_SIG, M, nullptr,nullptr,nullptr);
  gemm_wmma<128, 8,0,true ><<<GB,TB,0,stream>>>(t1,128,0, w_cc,b_cc, d_out,100,100, ACT_NONE, M, nullptr,nullptr,nullptr);
}
